// RGAT_66228395704802
// MI455X (gfx1250) — matrix operand order repaired and checked by reading, not compiled, not run on hardware
//
#include <hip/hip_runtime.h>
#include <cmath>

typedef __bf16 bf16_t;
typedef bf16_t v16bf __attribute__((ext_vector_type(16)));
typedef float  v8f   __attribute__((ext_vector_type(8)));

#define HIDC 512
#define DHC 128
#define HEADSC 4
#define RRELC 5
#define NEG_SLOPE_C 0.2f
#define EPS_C 1e-5f

// LDS B-chunk row stride in elements: 32 data + 8 pad (80 B, keeps 16B align, breaks bank cycles)
#define BROW 40

#if __has_builtin(__builtin_amdgcn_global_load_async_to_lds_b128) && \
    __has_builtin(__builtin_amdgcn_s_wait_asynccnt)
#define USE_ASYNC_LDS 1
// builtin prototype (from hipcc diagnostic): (v4i AS1*, v4i AS3*, imm int, imm int)
typedef int v4i_vs __attribute__((vector_size(16)));
typedef __attribute__((address_space(1))) v4i_vs* gas_p;
typedef __attribute__((address_space(3))) v4i_vs* las_p;
#else
#define USE_ASYNC_LDS 0
#endif

// ---- monotonic float <-> uint encoding for atomicMax-based segment max ----
__device__ __forceinline__ unsigned enc_f32(float f) {
  unsigned u = __float_as_uint(f);
  return (u & 0x80000000u) ? ~u : (u | 0x80000000u);
}
__device__ __forceinline__ float dec_f32(unsigned u) {
  unsigned v = (u & 0x80000000u) ? (u & 0x7FFFFFFFu) : ~u;
  return __uint_as_float(v);
}

// ---- stage one 32B sliver of the B chunk (per thread) into LDS ----
__device__ __forceinline__ void stage32(const bf16_t* __restrict__ g, bf16_t* l)
{
#if USE_ASYNC_LDS
  __builtin_amdgcn_global_load_async_to_lds_b128((gas_p)(void*)g,       (las_p)l,       0, 0);
  __builtin_amdgcn_global_load_async_to_lds_b128((gas_p)(void*)(g + 8), (las_p)(l + 8), 0, 0);
#else
  *(v16bf*)l = *(const v16bf*)g;
#endif
}

// ---------------------------------------------------------------------------
// GEMM: C[M x N] = A(f32, MxK row-major, cast->bf16) @ Bt(bf16, NxK row-major)
// Block = 256 threads = 8 waves; block tile 128(M) x 128(N); K-chunk 32.
// B chunk double-buffered in LDS (cooperatively staged, async path if
// available); each wave does 8x v_wmma_f32_16x16x32_bf16 per chunk from one
// A fragment + 8 LDS B fragments. Waves past M clamp (never exit: barriers).
// ---------------------------------------------------------------------------
__global__ __launch_bounds__(256) void gemm_a32_bw16(
    const float* __restrict__ A, const bf16_t* __restrict__ Bt,
    const float* __restrict__ bias, float* __restrict__ C,
    int M, int K, int ldc, int ncols)
{
  __shared__ bf16_t lsB[2][128 * BROW];   // 2 x 10,240 B

  int tid    = threadIdx.x;
  int wave   = tid >> 5;
  int mtiles = M >> 4;
  int mtile  = blockIdx.x * 8 + wave;
  if (mtile >= mtiles) mtile = mtiles - 1;   // clamp, keep wave for barriers
  int m0     = mtile * 16;
  int n_base = blockIdx.y * 128;
  int lane   = tid & 31;
  int l16    = lane & 15;
  int hi     = lane >> 4;

  // staging coordinates: 2 threads per output column row, 32B each
  int srow  = tid >> 1;        // 0..127  (column index within N tile)
  int shalf = tid & 1;         // 0/1     (first/second 16 K-values)
  const bf16_t* gB = Bt + (size_t)(n_base + srow) * K + shalf * 16;
  bf16_t* lB0 = &lsB[0][srow * BROW + shalf * 16];
  bf16_t* lB1 = &lsB[1][srow * BROW + shalf * 16];

  v8f acc[8];
  for (int t = 0; t < 8; ++t) {
    int col = n_base + t * 16 + l16;
    float bv = (bias != nullptr && col < ncols) ? bias[col] : 0.f;
    for (int r = 0; r < 8; ++r) acc[t][r] = bv;
  }

  const float* arow = A + (size_t)(m0 + l16) * K;

  // prologue: stage chunk 0 into buffer 0
  stage32(gB, lB0);
#if USE_ASYNC_LDS
  __builtin_amdgcn_s_wait_asynccnt(0);
#endif
  __syncthreads();

  int p = 0;
  for (int k0 = 0; k0 < K; k0 += 32) {
    // issue staging of the next chunk into the other buffer (overlaps WMMAs)
    if (k0 + 32 < K)
      stage32(gB + k0 + 32, p ? lB0 : lB1);

    // A fragment (f32 -> bf16 in regs):
    // lanes 0-15 rows M=0..15 hold K=0..15; lanes 16-31 K=16..31
    v8f alo = *(const v8f*)(arow + k0 + hi * 16);
    v8f ahf = *(const v8f*)(arow + k0 + hi * 16 + 8);
    v16bf af;
    #pragma unroll
    for (int i = 0; i < 8; ++i) { af[i] = (bf16_t)alo[i]; af[8 + i] = (bf16_t)ahf[i]; }

    const bf16_t* lbase = &lsB[p][0];
    #pragma unroll
    for (int t = 0; t < 8; ++t) {
      // B fragment: lane = column (l16), contiguous 16 K-values at hi*16
      v16bf bfv = *(const v16bf*)(lbase + (t * 16 + l16) * BROW + hi * 16);
      acc[t] = __builtin_amdgcn_wmma_f32_16x16x32_bf16(
          false, af, false, bfv, (short)0, acc[t], false, false);
    }

#if USE_ASYNC_LDS
    __builtin_amdgcn_s_wait_asynccnt(0);
#else
    asm volatile("" ::: "memory");
#endif
    __syncthreads();
    p ^= 1;
  }

  for (int t = 0; t < 8; ++t) {
    int col = n_base + t * 16 + l16;
    if (col >= ncols) continue;
    #pragma unroll
    for (int r = 0; r < 8; ++r)
      C[(size_t)(m0 + r + hi * 8) * ldc + col] = acc[t][r];
  }
}

// ---- transpose f32 [K x N] -> bf16 [Npad x K], zero padded rows ----
__global__ void transpose_to_bf16(const float* __restrict__ in, bf16_t* __restrict__ out,
                                  int K, int N, int Npad)
{
  int idx = blockIdx.x * blockDim.x + threadIdx.x;
  if (idx >= Npad * K) return;
  int n = idx / K, k = idx - n * K;
  out[(size_t)n * K + k] = (n < N) ? (bf16_t)in[(size_t)k * N + n] : (bf16_t)0.f;
}

// ---- per-node attention coefficients: a_s/a_d[n][h] = <h_row, a_vec> ----
__global__ void attn_coeff(const float* __restrict__ H,
                           const float* __restrict__ asrc, const float* __restrict__ adst,
                           float* __restrict__ a_s, float* __restrict__ a_d,
                           int n_nodes, int n_dst)
{
  int idx = blockIdx.x * blockDim.x + threadIdx.x;
  if (idx >= n_nodes * HEADSC) return;
  int n = idx >> 2, h = idx & 3;
  const float* row = H + (size_t)n * HIDC + h * DHC;
  const float* av = asrc + h * DHC;
  const float* dv = adst + h * DHC;
  float s = 0.f, d = 0.f;
  for (int i = 0; i < DHC; ++i) { float v = row[i]; s += v * av[i]; d += v * dv[i]; }
  a_s[idx] = s;
  if (n < n_dst) a_d[idx] = d;
}

__global__ void fill_u32(unsigned* __restrict__ p, unsigned v, int n)
{
  int i = blockIdx.x * blockDim.x + threadIdx.x;
  if (i < n) p[i] = v;
}

// ---- edge pass 1: leaky-relu score + segment max (encoded atomicMax) ----
__global__ void edge_score_max(const int* __restrict__ es, const int* __restrict__ ed,
                               const int* __restrict__ et, int rel,
                               const float* __restrict__ a_s, const float* __restrict__ a_d,
                               unsigned* __restrict__ smax, float* __restrict__ score, int E)
{
  int e = blockIdx.x * blockDim.x + threadIdx.x;
  if (e >= E || et[e] != rel) return;
  int s = es[e], d = ed[e];
  #pragma unroll
  for (int h = 0; h < HEADSC; ++h) {
    float v = a_s[s * 4 + h] + a_d[d * 4 + h];
    v = (v > 0.f) ? v : NEG_SLOPE_C * v;
    score[e * 4 + h] = v;
    atomicMax(&smax[d * 4 + h], enc_f32(v));
  }
}

// ---- edge pass 2: exp(score - max), segment sum ----
__global__ void edge_exp_sum(const int* __restrict__ ed, const int* __restrict__ et, int rel,
                             const unsigned* __restrict__ smax,
                             float* __restrict__ score, float* __restrict__ denom, int E)
{
  int e = blockIdx.x * blockDim.x + threadIdx.x;
  if (e >= E || et[e] != rel) return;
  int d = ed[e];
  #pragma unroll
  for (int h = 0; h < HEADSC; ++h) {
    float m = dec_f32(smax[d * 4 + h]);
    if (!__builtin_isfinite(m)) m = 0.f;
    float ev = __expf(score[e * 4 + h] - m);
    score[e * 4 + h] = ev;
    unsafeAtomicAdd(&denom[d * 4 + h], ev);
  }
}

// ---- edge pass 3: scatter alpha * h[src] into out[dst] (one block per edge) ----
__global__ void edge_aggregate(const int* __restrict__ es, const int* __restrict__ ed,
                               const int* __restrict__ et, int rel,
                               const float* __restrict__ score, const float* __restrict__ denom,
                               const float* __restrict__ H, float* __restrict__ out, int E)
{
  int e = blockIdx.x;
  if (et[e] != rel) return;
  int s = es[e], d = ed[e];
  __shared__ float alpha[HEADSC];
  if (threadIdx.x < HEADSC)
    alpha[threadIdx.x] = score[e * 4 + threadIdx.x] / fmaxf(denom[d * 4 + threadIdx.x], 1e-16f);
  __syncthreads();
  const float* hrow = H + (size_t)s * HIDC;
  float* orow = out + (size_t)d * HIDC;
  for (int c = threadIdx.x; c < HIDC; c += blockDim.x)
    unsafeAtomicAdd(&orow[c], alpha[c >> 7] * hrow[c]);
}

// ---- BatchNorm statistics: one block per channel, LDS tree reduction ----
__global__ void bn_stats(const float* __restrict__ X, int n_rows,
                         float* __restrict__ mu, float* __restrict__ var)
{
  __shared__ float ssum[256], ssq[256];
  int c = blockIdx.x;
  float s = 0.f, q = 0.f;
  for (int r = threadIdx.x; r < n_rows; r += 256) {
    float v = X[(size_t)r * HIDC + c];
    s += v; q += v * v;
  }
  ssum[threadIdx.x] = s; ssq[threadIdx.x] = q;
  __syncthreads();
  for (int off = 128; off > 0; off >>= 1) {
    if ((int)threadIdx.x < off) {
      ssum[threadIdx.x] += ssum[threadIdx.x + off];
      ssq[threadIdx.x]  += ssq[threadIdx.x + off];
    }
    __syncthreads();
  }
  if (threadIdx.x == 0) {
    float m = ssum[0] / (float)n_rows;
    mu[c] = m;
    var[c] = ssq[0] / (float)n_rows - m * m;
  }
}

// ---- BN apply + activation (mode 0 = ELU, 1 = ReLU), in place ----
__global__ void bn_apply(float* __restrict__ X, int n_rows,
                         const float* __restrict__ mu, const float* __restrict__ var,
                         const float* __restrict__ gamma, const float* __restrict__ beta,
                         int mode)
{
  int idx = blockIdx.x * blockDim.x + threadIdx.x;
  if (idx >= n_rows * HIDC) return;
  int c = idx & (HIDC - 1);
  float v = X[idx];
  float y = gamma[c] * ((v - mu[c]) * rsqrtf(var[c] + EPS_C)) + beta[c];
  if (mode == 0) y = (y > 0.f) ? y : (__expf(y) - 1.f);
  else           y = fmaxf(y, 0.f);
  X[idx] = y;
}

// ===========================================================================
extern "C" void kernel_launch(void* const* d_in, const int* in_sizes, int n_in,
                              void* d_out, int out_size, void* d_ws, size_t ws_size,
                              hipStream_t stream)
{
  const int N0 = 120000, E0 = 600000, E1 = 150000, DIN = 768;
  const int N1 = 30000, N2 = 8000, OUTC = 153, OUTPAD = 256;

  const float* x    = (const float*)d_in[0];
  const int*   es0  = (const int*)d_in[1];
  const int*   ed0  = (const int*)d_in[2];
  const int*   et0  = (const int*)d_in[3];
  const int*   es1  = (const int*)d_in[4];
  const int*   ed1  = (const int*)d_in[5];
  const int*   et1  = (const int*)d_in[6];
  const float* W0   = (const float*)d_in[7];
  const float* as0  = (const float*)d_in[8];
  const float* ad0  = (const float*)d_in[9];
  // d_in[10] b0: per-channel shift, cancels in training-mode BatchNorm
  const float* Wsk0 = (const float*)d_in[11];
  // d_in[12] bsk0: cancels in BatchNorm
  const float* g0   = (const float*)d_in[13];
  const float* be0  = (const float*)d_in[14];
  const float* W1   = (const float*)d_in[15];
  const float* as1  = (const float*)d_in[16];
  const float* ad1  = (const float*)d_in[17];
  // d_in[18] b1: cancels in BatchNorm
  const float* Wsk1 = (const float*)d_in[19];
  // d_in[20] bsk1: cancels in BatchNorm
  const float* g1   = (const float*)d_in[21];
  const float* be1  = (const float*)d_in[22];
  const float* Wm1  = (const float*)d_in[23];
  // d_in[24] bm1: cancels in BatchNorm
  const float* gm   = (const float*)d_in[25];
  const float* bmn  = (const float*)d_in[26];
  const float* Wm2  = (const float*)d_in[27];
  const float* bm2  = (const float*)d_in[28];

  // ---------------- workspace carve-out ----------------
  char* ws = (char*)d_ws;
  size_t off = 0;
  auto take = [&](size_t bytes) -> void* {
    void* p = ws + off;
    off = (off + bytes + 255) & ~(size_t)255;
    return p;
  };
  bf16_t* W0t   = (bf16_t*)take((size_t)RRELC * HIDC * DIN * 2);
  bf16_t* Wsk0t = (bf16_t*)take((size_t)HIDC * DIN * 2);
  bf16_t* W1t   = (bf16_t*)take((size_t)RRELC * HIDC * HIDC * 2);
  bf16_t* Wsk1t = (bf16_t*)take((size_t)HIDC * HIDC * 2);
  bf16_t* Wm1t  = (bf16_t*)take((size_t)HIDC * HIDC * 2);
  bf16_t* Wm2t  = (bf16_t*)take((size_t)OUTPAD * HIDC * 2);
  float* hbuf   = (float*)take((size_t)N0 * HIDC * 4);   // per-relation h = x @ W[j]
  float* out0   = (float*)take((size_t)N1 * HIDC * 4);
  float* out1   = (float*)take((size_t)N2 * HIDC * 4);
  float* zbuf   = (float*)take((size_t)N2 * HIDC * 4);
  float* a_sb   = (float*)take((size_t)N0 * HEADSC * 4);
  float* a_db   = (float*)take((size_t)N1 * HEADSC * 4);
  unsigned* smax= (unsigned*)take((size_t)N1 * HEADSC * 4);
  float* denom  = (float*)take((size_t)N1 * HEADSC * 4);
  float* score  = (float*)take((size_t)E0 * HEADSC * 4);
  float* bmu    = (float*)take((size_t)HIDC * 4);
  float* bvar   = (float*)take((size_t)HIDC * 4);

  // ---------------- weight transpose + bf16 convert ----------------
  auto tpose = [&](const float* in, bf16_t* outp, int K, int N, int Npad) {
    int tot = Npad * K;
    transpose_to_bf16<<<(tot + 255) / 256, 256, 0, stream>>>(in, outp, K, N, Npad);
  };
  for (int j = 0; j < RRELC; ++j)
    tpose(W0 + (size_t)j * DIN * HIDC, W0t + (size_t)j * HIDC * DIN, DIN, HIDC, HIDC);
  tpose(Wsk0, Wsk0t, DIN, HIDC, HIDC);
  for (int j = 0; j < RRELC; ++j)
    tpose(W1 + (size_t)j * HIDC * HIDC, W1t + (size_t)j * HIDC * HIDC, HIDC, HIDC, HIDC);
  tpose(Wsk1, Wsk1t, HIDC, HIDC, HIDC);
  tpose(Wm1,  Wm1t,  HIDC, HIDC, HIDC);
  tpose(Wm2,  Wm2t,  HIDC, OUTC, OUTPAD);

  auto gemm = [&](const float* A, const bf16_t* Bt, const float* bias, float* C,
                  int M, int K, int Npad, int ldc, int ncols) {
    dim3 g((M / 16 + 7) / 8, Npad / 128);
    gemm_a32_bw16<<<g, 256, 0, stream>>>(A, Bt, bias, C, M, K, ldc, ncols);
  };

  // ---------------- one RGAT layer ----------------
  auto layer = [&](const float* xin, int n_nodes, int n_dst, int K,
                   const int* es, const int* ed, const int* et, int E,
                   const bf16_t* Wt, const bf16_t* Wskt,
                   const float* asrc, const float* adst,
                   const float* gamma, const float* beta, float* outbuf) {
    // skip connection (bias cancels in BN)
    gemm(xin, Wskt, nullptr, outbuf, n_dst, K, HIDC, HIDC, HIDC);
    for (int j = 0; j < RRELC; ++j) {
      gemm(xin, Wt + (size_t)j * HIDC * K, nullptr, hbuf, n_nodes, K, HIDC, HIDC, HIDC);
      attn_coeff<<<(n_nodes * HEADSC + 255) / 256, 256, 0, stream>>>(
          hbuf, asrc + j * HEADSC * DHC, adst + j * HEADSC * DHC,
          a_sb, a_db, n_nodes, n_dst);
      fill_u32<<<(n_dst * HEADSC + 255) / 256, 256, 0, stream>>>(
          smax, 0x007FFFFFu /* enc(-inf) */, n_dst * HEADSC);
      fill_u32<<<(n_dst * HEADSC + 255) / 256, 256, 0, stream>>>(
          (unsigned*)denom, 0u, n_dst * HEADSC);
      edge_score_max<<<(E + 255) / 256, 256, 0, stream>>>(
          es, ed, et, j, a_sb, a_db, smax, score, E);
      edge_exp_sum<<<(E + 255) / 256, 256, 0, stream>>>(
          ed, et, j, smax, score, denom, E);
      edge_aggregate<<<E, 128, 0, stream>>>(
          es, ed, et, j, score, denom, hbuf, outbuf, E);
    }
    bn_stats<<<HIDC, 256, 0, stream>>>(outbuf, n_dst, bmu, bvar);
    bn_apply<<<(n_dst * HIDC + 255) / 256, 256, 0, stream>>>(
        outbuf, n_dst, bmu, bvar, gamma, beta, /*ELU*/0);
  };

  layer(x,    N0, N1, DIN,  es0, ed0, et0, E0, W0t, Wsk0t, as0, ad0, g0, be0, out0);
  layer(out0, N1, N2, HIDC, es1, ed1, et1, E1, W1t, Wsk1t, as1, ad1, g1, be1, out1);

  // ---------------- MLP head: Linear -> BN -> ReLU -> Linear ----------------
  gemm(out1, Wm1t, nullptr, zbuf, N2, HIDC, HIDC, HIDC, HIDC);
  bn_stats<<<HIDC, 256, 0, stream>>>(zbuf, N2, bmu, bvar);
  bn_apply<<<(N2 * HIDC + 255) / 256, 256, 0, stream>>>(
      zbuf, N2, bmu, bvar, gm, bmn, /*ReLU*/1);
  gemm(zbuf, Wm2t, bm2, (float*)d_out, N2, HIDC, OUTPAD, OUTC, OUTC);
}
